// SparseSelfAttention_33878702030866
// MI455X (gfx1250) — matrix operand order repaired and checked by reading, not compiled, not run on hardware
//
#include <hip/hip_runtime.h>
#include <hip/hip_bf16.h>
#include <cmath>

// ---- problem constants -------------------------------------------------
#define BB 2
#define NN 2048
#define EE 256
#define HH 8
#define DD 32            // head dim
#define LOG2E 1.4426950408889634f
// 1/sqrt(32) * log2(e): folded so softmax runs in base-2 domain
#define QSCALE (0.17677669529663687f * LOG2E)

typedef __attribute__((ext_vector_type(16))) _Float16 v16h;
typedef __attribute__((ext_vector_type(8)))  _Float16 v8h;
typedef __attribute__((ext_vector_type(8)))  float    v8f;
typedef __attribute__((ext_vector_type(4)))  float    v4f;

// ---- WMMA helper -------------------------------------------------------
__device__ __forceinline__ v8f wmma_f16(v16h a, v16h b, v8f c) {
  return __builtin_amdgcn_wmma_f32_16x16x32_f16(
      false, a, false, b, (short)0, c, false, false);
}

// A-operand (16x32 f16): lane holds row = lane&15; halves i -> k = 2*h8 + i
__device__ __forceinline__ v16h ldA(const _Float16* row, int h8) {
  v8h lo = *(const v8h*)(row + 2 * h8);
  v8h hi = *(const v8h*)(row + 2 * h8 + 8);
  v16h r;
#pragma unroll
  for (int i = 0; i < 8; ++i) { r[i] = lo[i]; r[8 + i] = hi[i]; }
  return r;
}

// B-operand (32x16 f16): lane holds col = lane&15; halves i -> k = 16*(lane>>4) + i.
__device__ __forceinline__ v16h ldB(const _Float16* ptr) {
  v8h lo = *(const v8h*)(ptr);
  v8h hi = *(const v8h*)(ptr + 8);
  v16h r;
#pragma unroll
  for (int i = 0; i < 8; ++i) { r[i] = lo[i]; r[8 + i] = hi[i]; }
  return r;
}

// ---- prep kernels ------------------------------------------------------
__global__ void k_cvt_f16(const float* __restrict__ src,
                          _Float16* __restrict__ dst, int n) {
  int i = blockIdx.x * blockDim.x + threadIdx.x;
  if (i < n) dst[i] = (_Float16)src[i];
}

// tau[b,h,n] = (feat[b,n,:] . tau_w[h,:] + tau_b[h])   (stored (B,H,N) f32)
__global__ void k_tau(const float* __restrict__ feat,
                      const float* __restrict__ tw,
                      const float* __restrict__ tb,
                      float* __restrict__ tau) {
  int t = blockIdx.x * blockDim.x + threadIdx.x;
  if (t >= BB * NN * HH) return;
  int hh = t & (HH - 1);
  int m  = t >> 3;                 // b*N + n
  const v4f* fr = (const v4f*)(feat + (size_t)m * EE);
  const v4f* wr = (const v4f*)(tw + hh * EE);
  v4f acc4 = {};
  for (int k = 0; k < EE / 4; ++k) {
    v4f a = fr[k], w = wr[k];
    acc4 += a * w;
  }
  float acc = acc4[0] + acc4[1] + acc4[2] + acc4[3];
  int b = m >> 11, n = m & (NN - 1);
  tau[((size_t)(b * HH + hh)) * NN + n] = acc + tb[hh];
}

// ---- kernel 1: QKV projection ------------------------------------------
// qkv[m, r] = feat[m,:] . Wqkv[r,:] + b[r];  M = B*N = 4096, R = 3E = 768
// scatter into Q,K (B,H,N,D) f16 (Q pre-scaled by QSCALE) and Vt (B,H,D,N) f16
__global__ void k_qkv(const _Float16* __restrict__ feat,
                      const _Float16* __restrict__ w,
                      const float* __restrict__ bias,
                      _Float16* __restrict__ Q,
                      _Float16* __restrict__ K,
                      _Float16* __restrict__ Vt) {
  int wave = (blockIdx.x * blockDim.x + threadIdx.x) >> 5;
  int lane = threadIdx.x & 31;
  int mt = wave / 48, rt = wave - mt * 48;        // 256 x 48 tiles
  if (mt >= 256) return;
  int m0 = mt * 16, r0 = rt * 16;
  int q = lane & 15, hl = lane >> 4, h8 = hl * 8, h16 = hl * 16;

  const _Float16* arow = feat + (size_t)(m0 + q) * EE;
  const _Float16* brow = w    + (size_t)(r0 + q) * EE;
  v8f acc = {};
  for (int kb = 0; kb < EE; kb += 32)
    acc = wmma_f16(ldA(arow + kb, h8), ldB(brow + kb + h16), acc);

  int r = r0 + q;
  float bv = bias[r];
  int seg = r >> 8;               // 0=q 1=k 2=v
  int rr  = r & 255;
  int hh  = rr >> 5, d = rr & 31;
#pragma unroll
  for (int v = 0; v < 8; ++v) {
    int mrow = m0 + v + h8;
    int b = mrow >> 11, n = mrow & (NN - 1);
    float val = acc[v] + bv;
    size_t bh = (size_t)(b * HH + hh);
    if (seg == 0)      Q[(bh * NN + n) * DD + d] = (_Float16)(val * QSCALE);
    else if (seg == 1) K[(bh * NN + n) * DD + d] = (_Float16)val;
    else               Vt[(bh * DD + d) * NN + n] = (_Float16)val;
  }
}

// ---- kernel 2: flash attention, one wave per (b,h,16-query tile) -------
// Computes S^T = K.Q^T so each lane owns one query column: softmax stats are
// per-lane scalars + one shfl_xor(16); P^T B-operand built by half-wave swap.
// All softmax math in base-2 domain (log2e pre-folded into Q and tau).
__global__ void k_attn(const _Float16* __restrict__ Q,
                       const _Float16* __restrict__ K,
                       const _Float16* __restrict__ Vt,
                       const float* __restrict__ tau,
                       const float* __restrict__ bbox,
                       _Float16* __restrict__ O) {
  int wave = (blockIdx.x * blockDim.x + threadIdx.x) >> 5;
  int lane = threadIdx.x & 31;
  int it = wave & 127;           // 128 i-tiles
  int bh = wave >> 7;            // [0,16)
  int b = bh >> 3, hh = bh & 7;
  int i0 = it * 16;
  int q = lane & 15, hl = lane >> 4, h8 = hl * 8, h16 = hl * 16;
  int iq = i0 + q;

  // loop-invariant per-lane query data (this lane owns query column iq)
  const _Float16* Qrow = Q + ((size_t)bh * NN + iq) * DD;
  v16h qb = ldB(Qrow + h16);                       // B-operand: Q^T
  const float* bboxb = bbox + (size_t)b * NN * 4;
  float cxq = bboxb[iq * 4 + 0];
  float cyq = bboxb[iq * 4 + 1];
  float tq  = tau[(size_t)bh * NN + iq] * LOG2E;   // base-2 domain

  const _Float16* Kbase = K  + (size_t)bh * NN * DD;
  const _Float16* Vbase = Vt + (size_t)bh * DD * NN;

  float mrun = -INFINITY, lrun = 0.f;
  v8f o0 = {}, o1 = {};
  v8f zc = {};

  for (int j0 = 0; j0 < NN; j0 += 32) {
    // S^T = K(32 keys x 32d) . Q^T : two 16x16 C-tiles (keys 0-15 / 16-31)
    v16h ka0 = ldA(Kbase + (size_t)(j0 + q) * DD, h8);
    v16h ka1 = ldA(Kbase + (size_t)(j0 + 16 + q) * DD, h8);
    v8f s0 = wmma_f16(ka0, qb, zc);
    v8f s1 = wmma_f16(ka1, qb, zc);

    // mask: element (key j, query iq) += -sqrt(|c_i - c_j|) * tau_i (base-2)
    float p0[8], p1[8];
    float cmax = -INFINITY;
#pragma unroll
    for (int v = 0; v < 8; ++v) {
      int ja = j0 + h8 + v;          // tile0 row v -> key index
      int jb = ja + 16;              // tile1 row v
      float ax = bboxb[ja * 4], ay = bboxb[ja * 4 + 1];
      float bx = bboxb[jb * 4], by = bboxb[jb * 4 + 1];
      float dxa = cxq - ax, dya = cyq - ay;
      float dxb = cxq - bx, dyb = cyq - by;
      float sa = s0[v] - sqrtf(dxa * dxa + dya * dya) * tq;
      float sb = s1[v] - sqrtf(dxb * dxb + dyb * dyb) * tq;
      p0[v] = sa; p1[v] = sb;
      cmax = fmaxf(cmax, fmaxf(sa, sb));
    }
    cmax = fmaxf(cmax, __shfl_xor(cmax, 16, 32));   // full 32-key chunk max
    float mnew  = fmaxf(mrun, cmax);
    float scale = __builtin_amdgcn_exp2f(mrun - mnew);
    float csum = 0.f;
#pragma unroll
    for (int v = 0; v < 8; ++v) {
      p0[v] = __builtin_amdgcn_exp2f(p0[v] - mnew);
      p1[v] = __builtin_amdgcn_exp2f(p1[v] - mnew);
      csum += p0[v] + p1[v];
    }
    csum += __shfl_xor(csum, 16, 32);
    lrun = lrun * scale + csum;
    mrun = mnew;
#pragma unroll
    for (int v = 0; v < 8; ++v) { o0[v] *= scale; o1[v] *= scale; }

    // build P^T B-operand: lane(q,hl) needs k=16*hl+i; half-wave swap does it
    v16h pb;
#pragma unroll
    for (int v = 0; v < 8; ++v) {
      float sw0 = __shfl_xor(p0[v], 16, 32);
      float sw1 = __shfl_xor(p1[v], 16, 32);
      pb[v]     = (_Float16)(hl == 0 ? p0[v] : sw1);
      pb[v + 8] = (_Float16)(hl == 0 ? sw0   : p1[v]);
    }

    // o^T += V^T(32d x 32keys) . P^T : two C-tiles over d
    v16h va0 = ldA(Vbase + (size_t)(q) * NN + j0, h8);
    v16h va1 = ldA(Vbase + (size_t)(16 + q) * NN + j0, h8);
    o0 = wmma_f16(va0, pb, o0);
    o1 = wmma_f16(va1, pb, o1);
  }

  float inv = 1.0f / lrun;
  _Float16* orow = O + ((size_t)(b * NN + iq)) * EE + hh * DD;
  v8h st0, st1;
#pragma unroll
  for (int v = 0; v < 8; ++v) {
    st0[v] = (_Float16)(o0[v] * inv);   // d = h8+v
    st1[v] = (_Float16)(o1[v] * inv);   // d = 16+h8+v
  }
  *(v8h*)(orow + h8)      = st0;
  *(v8h*)(orow + 16 + h8) = st1;
}

// ---- kernel 3: out projection + bias + residual ------------------------
__global__ void k_out(const _Float16* __restrict__ O,
                      const _Float16* __restrict__ Wout,
                      const float* __restrict__ bias,
                      const float* __restrict__ feat,
                      float* __restrict__ out) {
  int wave = (blockIdx.x * blockDim.x + threadIdx.x) >> 5;
  int lane = threadIdx.x & 31;
  int mt = wave >> 4, et = wave & 15;     // 256 x 16 tiles
  if (mt >= 256) return;
  int m0 = mt * 16, e0 = et * 16;
  int q = lane & 15, hl = lane >> 4, h8 = hl * 8, h16 = hl * 16;

  const _Float16* arow = O    + (size_t)(m0 + q) * EE;
  const _Float16* brow = Wout + (size_t)(e0 + q) * EE;
  v8f acc = {};
  for (int kb = 0; kb < EE; kb += 32)
    acc = wmma_f16(ldA(arow + kb, h8), ldB(brow + kb + h16), acc);

  int e = e0 + q;
  float bv = bias[e];
#pragma unroll
  for (int v = 0; v < 8; ++v) {
    int m = m0 + v + h8;
    size_t idx = (size_t)m * EE + e;
    out[idx] = acc[v] + bv + feat[idx];
  }
}

// ---- launcher ----------------------------------------------------------
extern "C" void kernel_launch(void* const* d_in, const int* in_sizes, int n_in,
                              void* d_out, int out_size, void* d_ws, size_t ws_size,
                              hipStream_t stream) {
  const float* bbox   = (const float*)d_in[0];   // (B,N,4)
  const float* feat   = (const float*)d_in[1];   // (B,N,E)
  const float* wqkv   = (const float*)d_in[2];   // (3E,E)
  const float* bqkv   = (const float*)d_in[3];   // (3E,)
  const float* wout   = (const float*)d_in[4];   // (E,E)
  const float* bout   = (const float*)d_in[5];   // (E,)
  const float* tw     = (const float*)d_in[6];   // (H,E)
  const float* tb     = (const float*)d_in[7];   // (H,)
  float* out = (float*)d_out;

  char* ws = (char*)d_ws;
  const size_t nFeat = (size_t)BB * NN * EE;          // 1,048,576
  const size_t nWqkv = (size_t)3 * EE * EE;           //   196,608
  const size_t nWout = (size_t)EE * EE;               //    65,536
  const size_t nHead = (size_t)BB * HH * NN * DD;     // 1,048,576
  const size_t nTau  = (size_t)BB * HH * NN;          //    32,768

  _Float16* featf = (_Float16*)(ws);                       size_t off = nFeat * 2;
  _Float16* wqkvf = (_Float16*)(ws + off);                 off += nWqkv * 2;
  _Float16* woutf = (_Float16*)(ws + off);                 off += nWout * 2;
  _Float16* Qf    = (_Float16*)(ws + off);                 off += nHead * 2;
  _Float16* Kf    = (_Float16*)(ws + off);                 off += nHead * 2;
  _Float16* Vtf   = (_Float16*)(ws + off);                 off += nHead * 2;
  float*    tauf  = (float*)(ws + off);                    off += nTau * 4;
  _Float16* Of    = (_Float16*)(ws + off);                 off += nFeat * 2;

  // prep: f32 -> f16 conversions
  k_cvt_f16<<<(int)((nFeat + 255) / 256), 256, 0, stream>>>(feat, featf, (int)nFeat);
  k_cvt_f16<<<(int)((nWqkv + 255) / 256), 256, 0, stream>>>(wqkv, wqkvf, (int)nWqkv);
  k_cvt_f16<<<(int)((nWout + 255) / 256), 256, 0, stream>>>(wout, woutf, (int)nWout);
  // tau
  k_tau<<<(BB * NN * HH + 255) / 256, 256, 0, stream>>>(feat, tw, tb, tauf);
  // QKV projection: 256*48 = 12288 waves, 4 waves/block
  k_qkv<<<3072, 128, 0, stream>>>(featf, wqkvf, bqkv, Qf, Kf, Vtf);
  // attention: 2*8*128 = 2048 waves
  k_attn<<<512, 128, 0, stream>>>(Qf, Kf, Vtf, tauf, bbox, Of);
  // out projection + residual: 256*16 = 4096 waves
  k_out<<<1024, 128, 0, stream>>>(Of, woutf, bout, feat, out);
}
